// FmoeCatEmbedFeedForward_1743756722201
// MI455X (gfx1250) — matrix operand order repaired and checked by reading, not compiled, not run on hardware
//
#include <hip/hip_runtime.h>
#include <hip/hip_bf16.h>

// ---------------------------------------------------------------------------
// MoE (top-1) forward for MI455X / gfx1250, wave32 + WMMA bf16.
//
// B=16 S=2048 D=512 EDIM=256 H=1024 NE=4 -> N=32768 tokens.
// Selected-expert FFN = 68.7 GFLOP over ~170MB HBM traffic -> memory bound
// (~7.4us @ 23.3TB/s) once the math runs on v_wmma_f32_16x16x32_bf16.
// bf16 expert weights (8MB total) stay L2-resident (192MB L2).
//
// Round-2 change: TILE_M 16 -> 32. Each B fragment (1KB from a weight row)
// now feeds TWO wmma ops (two M-subtiles), halving L2 weight traffic and
// the vmem:wmma ratio. LDS/workgroup: Xs 32KB + Hs 64KB = 96KB (< 320KB).
// ---------------------------------------------------------------------------

#define NTOK   32768          // B*S
#define DIN    512
#define EDIMC  256
#define HID    1024
#define NEXP   4
#define RDIM   (DIN + EDIMC)  // 768
#define TILE_M 32
#define EPSF   1e-10f

typedef __attribute__((ext_vector_type(16))) __bf16 v16bf;
typedef __attribute__((ext_vector_type(8)))  float  v8f;

union Frag { v16bf v; uint4 q[2]; };

__device__ __forceinline__ unsigned short f2bf(float f) {
  unsigned int u = __float_as_uint(f);
  u += 0x7fffu + ((u >> 16) & 1u);   // round-to-nearest-even
  return (unsigned short)(u >> 16);
}

// ---------------------------------------------------------------------------
// 0) zero the atomic cells used this launch (harness does not re-poison)
// ---------------------------------------------------------------------------
__global__ void moe_init_kernel(int* counts, float* l1sum, float* imp) {
  int t = threadIdx.x;
  if (t < NEXP) { counts[t] = 0; imp[t] = 0.0f; }
  if (t == 0)   { *l1sum = 0.0f; }
}

// ---------------------------------------------------------------------------
// 1) fp32 -> bf16 weight conversion (once per launch; 4.2M elems total)
// ---------------------------------------------------------------------------
__global__ void moe_cvt_kernel(const float* __restrict__ src,
                               unsigned short* __restrict__ dst, int n) {
  int i = blockIdx.x * blockDim.x + threadIdx.x;
  int stride = gridDim.x * blockDim.x;
  for (; i < n; i += stride) dst[i] = f2bf(src[i]);
}

// ---------------------------------------------------------------------------
// 2) router: logits -> softmax -> top-1, per-expert token lists, loss sums.
// ---------------------------------------------------------------------------
__global__ __launch_bounds__(256) void moe_router_kernel(
    const float* __restrict__ x, const float* __restrict__ emb,
    const float* __restrict__ Wr,
    int* __restrict__ counts, int* __restrict__ lists,
    float* __restrict__ gatev, float* __restrict__ l1sum,
    float* __restrict__ imp) {
  __shared__ float Ws[RDIM * NEXP];
  int tid = threadIdx.x;
  for (int i = tid; i < RDIM * NEXP; i += blockDim.x) Ws[i] = Wr[i];
  __syncthreads();

  int tok = blockIdx.x * blockDim.x + tid;
  if (tok >= NTOK) return;

  float a0 = 0.f, a1 = 0.f, a2 = 0.f, a3 = 0.f;
  const float* er = emb + (size_t)tok * EDIMC;     // cat([embed, inputs])
  for (int k = 0; k < EDIMC; ++k) {
    float v = er[k];
    const float* w = Ws + k * NEXP;
    a0 += v * w[0]; a1 += v * w[1]; a2 += v * w[2]; a3 += v * w[3];
  }
  const float* xr = x + (size_t)tok * DIN;
  for (int k = 0; k < DIN; ++k) {
    float v = xr[k];
    const float* w = Ws + (EDIMC + k) * NEXP;
    a0 += v * w[0]; a1 += v * w[1]; a2 += v * w[2]; a3 += v * w[3];
  }
  // softmax over 4
  float mx = fmaxf(fmaxf(a0, a1), fmaxf(a2, a3));
  float e0 = __expf(a0 - mx), e1 = __expf(a1 - mx);
  float e2 = __expf(a2 - mx), e3 = __expf(a3 - mx);
  float se = e0 + e1 + e2 + e3;
  float rs = 1.0f / se;
  float p0 = e0 * rs, p1 = e1 * rs, p2 = e2 * rs, p3 = e3 * rs;
  // argmax (first-max tie-break like jnp.argmax)
  int   gi = 0;  float gv = p0;
  if (p1 > gv) { gv = p1; gi = 1; }
  if (p2 > gv) { gv = p2; gi = 2; }
  if (p3 > gv) { gv = p3; gi = 3; }
  // SparseL1Loss term: sum(p)/(||p||_2 + eps)
  float sump = p0 + p1 + p2 + p3;
  float nrm  = sqrtf(p0 * p0 + p1 * p1 + p2 * p2 + p3 * p3);
  atomicAdd(l1sum, sump / (nrm + EPSF));
  atomicAdd(&imp[0], p0); atomicAdd(&imp[1], p1);
  atomicAdd(&imp[2], p2); atomicAdd(&imp[3], p3);
  int pos = atomicAdd(&counts[gi], 1);
  lists[gi * NTOK + pos] = tok;
  gatev[tok] = gv;
}

// ---------------------------------------------------------------------------
// WMMA fragment loaders, per CDNA5 ISA 7.12.2 (wave32) layouts.
// A (16x32 bf16): m=lane&15; lanes<16 hold K 0..7 & 16..23, lanes>=16 hold
//                 K 8..15 & 24..31  -> two 16B LDS chunks per lane.
// B (32x16 bf16): n=lane&15; lanes<16 hold K 0..7 & 16..23, lanes>=16 hold
//                 K 8..15 & 24..31 -> two 16B chunks of one weight row.
// C/D (16x16 f32): VGPR r -> M = r + (lane<16 ? 0 : 8), N = lane&15.
// ---------------------------------------------------------------------------
__device__ __forceinline__ v16bf loadA_lds(const unsigned short* base,
                                           int stride, int k0, int lane) {
  int m  = lane & 15;
  int kb = (lane < 16) ? 0 : 8;
  const unsigned short* p = base + m * stride + k0 + kb;
  Frag f;
  f.q[0] = *(const uint4*)(p);
  f.q[1] = *(const uint4*)(p + 16);
  return f.v;
}

__device__ __forceinline__ v16bf loadB_glb(const unsigned short* __restrict__ w,
                                           int row0, int k0, int K, int lane) {
  int n  = lane & 15;
  int kb = k0 + ((lane < 16) ? 0 : 8);
  const uint4* p = (const uint4*)(w + (size_t)(row0 + n) * K + kb);
  Frag f;
  f.q[0] = p[0];
  f.q[1] = p[2];
  return f.v;
}

// ---------------------------------------------------------------------------
// 3) expert FFN: one block = one expert-tile of 32 tokens (2 M-subtiles).
//    Phase 1: H = relu(X @ W1^T + b1)  (K=512,  64 N-tiles over H=1024)
//    Phase 2: Y = H @ W2^T + b2        (K=1024, 32 N-tiles over D=512)
//    8 waves/block, N-tiles round-robined across waves; each B fragment is
//    reused by both M-subtiles (2 wmma per 32B/lane weight load).
// ---------------------------------------------------------------------------
__global__ __launch_bounds__(256, 1) void moe_ffn_kernel(
    const float* __restrict__ x,
    const unsigned short* __restrict__ w1b, const float* __restrict__ b1,
    const unsigned short* __restrict__ w2b, const float* __restrict__ b2,
    const int* __restrict__ counts, const int* __restrict__ lists,
    const float* __restrict__ gatev, float* __restrict__ out) {
  __shared__ unsigned short Xs[TILE_M * DIN];   // 32 KB
  __shared__ unsigned short Hs[TILE_M * HID];   // 64 KB
  __shared__ int   toks[TILE_M];
  __shared__ float gvs[TILE_M];
  __shared__ int s_e, s_t;

  int tid = threadIdx.x;

  // resolve (expert, tile) from per-expert counts; block-uniform.
  if (tid == 0) {
    int b = blockIdx.x, e = -1, t = -1;
    for (int i = 0; i < NEXP; ++i) {
      int nt = (counts[i] + TILE_M - 1) / TILE_M;
      if (b < nt) { e = i; t = b; break; }
      b -= nt;
    }
    s_e = e; s_t = t;
  }
  __syncthreads();
  int e = s_e, t = s_t;
  if (e < 0) return;                       // uniform across the block
  int cnt = counts[e];

  if (tid < TILE_M) {
    int idx = t * TILE_M + tid;
    int tok = (idx < cnt) ? lists[e * NTOK + idx] : -1;
    toks[tid] = tok;
    gvs[tid]  = (tok >= 0) ? gatev[tok] : 0.0f;
  }
  __syncthreads();

  // gather 32 token rows of x -> bf16 in LDS (pad rows with zero)
  {
    int row = tid >> 3;                    // 0..31 (8 threads/row)
    int c0  = (tid & 7) * 64;              // 64 contiguous floats each
    int tok = toks[row];
    if (tok >= 0) {
      const float* xr = x + (size_t)tok * DIN + c0;
      for (int i = 0; i < 64; i += 4) {
        float4 f = *(const float4*)(xr + i);
        Xs[row * DIN + c0 + i + 0] = f2bf(f.x);
        Xs[row * DIN + c0 + i + 1] = f2bf(f.y);
        Xs[row * DIN + c0 + i + 2] = f2bf(f.z);
        Xs[row * DIN + c0 + i + 3] = f2bf(f.w);
      }
    } else {
      for (int i = 0; i < 64; ++i) Xs[row * DIN + c0 + i] = 0;
    }
  }
  __syncthreads();

  int lane = tid & 31;
  int wv   = tid >> 5;                     // wave 0..7
  int n    = lane & 15;
  int mrow = (lane < 16) ? 0 : 8;

  // ---- phase 1: H = relu(X @ W1^T + b1) ----
  const unsigned short* W1e = w1b + (size_t)e * HID * DIN;
  const float* b1e = b1 + e * HID;
  for (int ntile = wv; ntile < HID / 16; ntile += 8) {
    int h0 = ntile * 16;
    v8f c0 = {};                           // rows 0..15
    v8f c1 = {};                           // rows 16..31
    for (int kt = 0; kt < DIN / 32; ++kt) {
      v16bf B  = loadB_glb(W1e, h0, kt * 32, DIN, lane);
      v16bf A0 = loadA_lds(Xs,            DIN, kt * 32, lane);
      v16bf A1 = loadA_lds(Xs + 16 * DIN, DIN, kt * 32, lane);
      c0 = __builtin_amdgcn_wmma_f32_16x16x32_bf16(false, A0, false, B,
                                                   (short)0, c0, false, false);
      c1 = __builtin_amdgcn_wmma_f32_16x16x32_bf16(false, A1, false, B,
                                                   (short)0, c1, false, false);
    }
    float bias = b1e[h0 + n];
#pragma unroll
    for (int r = 0; r < 8; ++r) {
      float v0 = c0[r] + bias;
      float v1 = c1[r] + bias;
      v0 = v0 > 0.0f ? v0 : 0.0f;
      v1 = v1 > 0.0f ? v1 : 0.0f;
      Hs[(mrow + r) * HID + h0 + n]        = f2bf(v0);
      Hs[(16 + mrow + r) * HID + h0 + n]   = f2bf(v1);
    }
  }
  __syncthreads();

  // ---- phase 2: Y = H @ W2^T + b2, gate-scaled scatter store ----
  const unsigned short* W2e = w2b + (size_t)e * DIN * HID;
  const float* b2e = b2 + e * DIN;
  for (int ntile = wv; ntile < DIN / 16; ntile += 8) {
    int d0 = ntile * 16;
    v8f c0 = {};
    v8f c1 = {};
    for (int kt = 0; kt < HID / 32; ++kt) {
      v16bf B  = loadB_glb(W2e, d0, kt * 32, HID, lane);
      v16bf A0 = loadA_lds(Hs,            HID, kt * 32, lane);
      v16bf A1 = loadA_lds(Hs + 16 * HID, HID, kt * 32, lane);
      c0 = __builtin_amdgcn_wmma_f32_16x16x32_bf16(false, A0, false, B,
                                                   (short)0, c0, false, false);
      c1 = __builtin_amdgcn_wmma_f32_16x16x32_bf16(false, A1, false, B,
                                                   (short)0, c1, false, false);
    }
    float bias = b2e[d0 + n];
#pragma unroll
    for (int r = 0; r < 8; ++r) {
      int m0 = mrow + r;
      int m1 = 16 + mrow + r;
      int tok0 = toks[m0];
      int tok1 = toks[m1];
      if (tok0 >= 0)
        out[(size_t)tok0 * DIN + d0 + n] = (c0[r] + bias) * gvs[m0];
      if (tok1 >= 0)
        out[(size_t)tok1 * DIN + d0 + n] = (c1[r] + bias) * gvs[m1];
    }
  }
}

// ---------------------------------------------------------------------------
// 4) finalize the two scalar losses
// ---------------------------------------------------------------------------
__global__ void moe_loss_kernel(const float* __restrict__ l1sum,
                                const float* __restrict__ imp,
                                float* __restrict__ out_tail) {
  if (threadIdx.x == 0) {
    float l1 = *l1sum / (float)NTOK;
    float m  = (imp[0] + imp[1] + imp[2] + imp[3]) * 0.25f;
    float v  = 0.0f;
    for (int j = 0; j < NEXP; ++j) { float d = imp[j] - m; v += d * d; }
    v *= 0.25f;                            // population variance (ddof=0)
    out_tail[0] = l1;
    out_tail[1] = v / (m * m + EPSF);
  }
}

// ---------------------------------------------------------------------------
extern "C" void kernel_launch(void* const* d_in, const int* in_sizes, int n_in,
                              void* d_out, int out_size, void* d_ws,
                              size_t ws_size, hipStream_t stream) {
  const float* x   = (const float*)d_in[0];  // [B,S,D]
  const float* emb = (const float*)d_in[1];  // [B,S,EDIM]
  const float* Wr  = (const float*)d_in[2];  // [D+EDIM, NE]
  const float* w1  = (const float*)d_in[3];  // [NE,H,D]
  const float* b1  = (const float*)d_in[4];  // [NE,H]
  const float* w2  = (const float*)d_in[5];  // [NE,D,H]
  const float* b2  = (const float*)d_in[6];  // [NE,D]
  float* out = (float*)d_out;                // [N*D] ++ [l1, imp]

  // workspace layout (all chunks naturally aligned)
  char* ws = (char*)d_ws;
  unsigned short* w1b = (unsigned short*)ws;  ws += (size_t)NEXP * HID * DIN * 2;
  unsigned short* w2b = (unsigned short*)ws;  ws += (size_t)NEXP * DIN * HID * 2;
  int*   lists  = (int*)ws;                   ws += (size_t)NEXP * NTOK * 4;
  float* gatev  = (float*)ws;                 ws += (size_t)NTOK * 4;
  int*   counts = (int*)ws;                   ws += NEXP * 4;
  float* l1sum  = (float*)ws;                 ws += 4;
  float* imp    = (float*)ws;                 ws += NEXP * 4;

  moe_init_kernel<<<1, 32, 0, stream>>>(counts, l1sum, imp);

  const int ncvt = NEXP * HID * DIN;          // 2,097,152 per weight tensor
  moe_cvt_kernel<<<2048, 256, 0, stream>>>(w1, w1b, ncvt);
  moe_cvt_kernel<<<2048, 256, 0, stream>>>(w2, w2b, ncvt);

  moe_router_kernel<<<NTOK / 256, 256, 0, stream>>>(
      x, emb, Wr, counts, lists, gatev, l1sum, imp);

  moe_ffn_kernel<<<NTOK / TILE_M + NEXP, 256, 0, stream>>>(
      x, w1b, b1, w2b, b2, counts, lists, gatev, out);

  moe_loss_kernel<<<1, 32, 0, stream>>>(l1sum, imp, out + (size_t)NTOK * DIN);
}
